// LODE_T_77945066488240
// MI455X (gfx1250) — matrix operand order repaired and checked by reading, not compiled, not run on hardware
//
#include <hip/hip_runtime.h>

// ---------------------------------------------------------------------------
// LODE torus dynamics (d=2, NEURONS=256) for MI455X / gfx1250.
// Forward-mode JVP formulation: 3 streams (value + 2 tangents) through the
// 256x256 trunk layer via v_wmma_f32_16x16x32_f16; everything else f32 VALU.
// S_BLK=32 -> ~105 KB LDS -> 2 workgroups/WGP -> 4 waves/SIMD for latency
// hiding; sched_barrier pins DS-load bursts ahead of WMMA bursts per k-step.
// ---------------------------------------------------------------------------

typedef __attribute__((ext_vector_type(16))) _Float16 v16h;
typedef __attribute__((ext_vector_type(8)))  float    v8f;

#define NEU     256
#define S_BLK   32              // samples per block
#define ROWS    96              // 3 * S_BLK (h0, dh0_q0, dh0_q1)
#define MT_CNT  6               // ROWS / 16
#define STRIDE  264             // NEU + 8 halves padding (row = 528 B)
#define EPSV    1e-4f

union FragU {
    uint4 u4[2];
    v16h  h;
};

// Pre-swizzle W1 (f32 256x256 row-major, [k][n]) into f16 dense-B fragment
// order: for (nt, ks, lane, v): value = W1[ks*32 + (lane>=16)*16 + 2v + {0,1}]
// [nt*16 + (lane&15)].  Each lane's 8 dwords are contiguous -> coalesced 32B.
__global__ void prep_w1_swizzle(const float* __restrict__ W1,
                                _Float16* __restrict__ W1sw) {
    int o = blockIdx.x * blockDim.x + threadIdx.x;   // 65536 dwords total
    if (o >= 65536) return;
    int v    = o & 7;
    int lane = (o >> 3) & 31;
    int ks   = (o >> 8) & 7;
    int nt   = o >> 11;
    int ncol = nt * 16 + (lane & 15);
    int k    = ks * 32 + ((lane >> 4) << 4) + 2 * v;
    W1sw[2 * o]     = (_Float16)W1[k * NEU + ncol];
    W1sw[2 * o + 1] = (_Float16)W1[(k + 1) * NEU + ncol];
}

__device__ __forceinline__ float softplusf(float z) {
    return (z > 20.f) ? z : log1pf(expf(z));
}

__device__ __forceinline__ void sched_fence() {
#if __has_builtin(__builtin_amdgcn_sched_barrier)
    __builtin_amdgcn_sched_barrier(0);
#endif
}

__global__ __launch_bounds__(256, 2)
void lode_dyn_kernel(const float* __restrict__ x,
                     const float* __restrict__ W0, const float* __restrict__ b0,
                     const _Float16* __restrict__ W1sw, const float* __restrict__ b1,
                     const float* __restrict__ Wd, const float* __restrict__ bd,
                     const float* __restrict__ Wo, const float* __restrict__ bo,
                     const float* __restrict__ Wv,
                     float* __restrict__ out, int n) {
    extern __shared__ char smem[];
    _Float16* A    = (_Float16*)smem;                  // ROWS x STRIDE (layer-1 input)
    _Float16* Aout = A + ROWS * STRIDE;                // ROWS x STRIDE (layer-1 output)
    float* hWd0 = (float*)(Aout + ROWS * STRIDE);      // head weights, f32
    float* hWd1 = hWd0 + NEU;
    float* hWo  = hWd1 + NEU;
    float* hWv  = hWo  + NEU;

    const int tid = threadIdx.x;

    // stage head weights into LDS (256 threads, one neuron each)
    hWd0[tid] = Wd[tid * 2 + 0];
    hWd1[tid] = Wd[tid * 2 + 1];
    hWo[tid]  = Wo[tid];
    hWv[tid]  = Wv[tid];

    const int s  = tid >> 3;          // sample within block (0..31)
    const int p  = tid & 7;           // eighth (stage A: 32 neurons; stage C: dot part)
    const int gs = blockIdx.x * S_BLK + s;
    const bool valid = gs < n;

    float q0 = 0.f, q1 = 0.f, qd0 = 0.f, qd1 = 0.f, tau0 = 0.f, tau1 = 0.f;
    if (valid) {
        q0   = x[gs * 6 + 0]; q1   = x[gs * 6 + 1];
        qd0  = x[gs * 6 + 2]; qd1  = x[gs * 6 + 3];
        tau0 = x[gs * 6 + 4]; tau1 = x[gs * 6 + 5];
    }
    const float c0 = cosf(q0), s0 = sinf(q0), c1 = cosf(q1), s1 = sinf(q1);

    // ---- Stage A: layer 0 (K=4, VALU f32) + tanh + tangent seeding --------
    // emb = (c0, c1, s0, s1); t0 = (-s0, 0, c0, 0); t1 = (0, -s1, 0, c1)
    for (int j = 0; j < 32; ++j) {
        const int nn = (p << 5) + j;
        const float w0n = W0[nn];
        const float w1n = W0[NEU + nn];
        const float w2n = W0[2 * NEU + nn];
        const float w3n = W0[3 * NEU + nn];
        float z = fmaf(c0, w0n, fmaf(c1, w1n, fmaf(s0, w2n, fmaf(s1, w3n, b0[nn]))));
        float h = tanhf(z);
        float u = 1.f - h * h;                         // tanh'
        A[s * STRIDE + nn]               = (_Float16)h;
        A[(S_BLK + s) * STRIDE + nn]     = (_Float16)(u * fmaf(-s0, w0n, c0 * w2n));
        A[(2 * S_BLK + s) * STRIDE + nn] = (_Float16)(u * fmaf(-s1, w1n, c1 * w3n));
    }
    __syncthreads();

    // ---- Stage B: layer 1 GEMM (96 x 256 x 256) via WMMA f16 --------------
    const int wave = tid >> 5;
    const int lane = tid & 31;
    const int lhi  = lane >> 4;       // lane-half (K chunk select)
    const int lcol = lane & 15;       // N column within tile / A row within tile

    for (int t = 0; t < 2; ++t) {
        const int nt = wave + t * 8;  // this wave's N-tile
        v8f acc[MT_CNT];
#pragma unroll
        for (int mt = 0; mt < MT_CNT; ++mt)
            acc[mt] = (v8f){0.f,0.f,0.f,0.f,0.f,0.f,0.f,0.f};

#pragma unroll
        for (int ks = 0; ks < 8; ++ks) {
            // B fragment: coalesced 32B per lane from pre-swizzled buffer
            FragU bf;
            const uint4* bp = (const uint4*)(W1sw + (size_t)(((nt * 8 + ks) * 32 + lane) * 16));
            bf.u4[0] = bp[0];
            bf.u4[1] = bp[1];
            // A fragments: contiguous K order matching B -> lane-half picks 16-wide
            // K chunk: lanes 0-15: K=[0,16); lanes 16-31: K=[16,32).
            const int koff = ks * 32 + (lhi << 4);     // in halves
            FragU af[MT_CNT];
#pragma unroll
            for (int mt = 0; mt < MT_CNT; ++mt) {
                const uint4* ap = (const uint4*)(A + (mt * 16 + lcol) * STRIDE + koff);
                af[mt].u4[0] = ap[0];                  // K chunk 0 (8 halves)
                af[mt].u4[1] = ap[1];                  // K chunk +8 halves
            }
            // Pin all fragment loads before the WMMA burst so dscnt waits are
            // partial and remaining loads overlap the matrix pipe.
            sched_fence();
#pragma unroll
            for (int mt = 0; mt < MT_CNT; ++mt)
                acc[mt] = __builtin_amdgcn_wmma_f32_16x16x32_f16(
                    false, af[mt].h, false, bf.h, (short)0, acc[mt], false, false);
            sched_fence();
        }

        // Postprocess: bias+tanh for value stream, tanh'-scale for tangents.
        // Tangent row 32+s / 64+s maps to the SAME lane/reg slot in tile
        // mt-2 / mt-4, so no cross-lane traffic is needed.
        const int ncol = nt * 16 + lcol;
        const int rof  = lhi * 8;                      // C-layout M offset
        const float b1v = b1[ncol];
        float th[2][8];
#pragma unroll
        for (int mt = 0; mt < 2; ++mt)
#pragma unroll
            for (int r = 0; r < 8; ++r) {
                float h = tanhf(acc[mt][r] + b1v);
                th[mt][r] = 1.f - h * h;
                Aout[(mt * 16 + rof + r) * STRIDE + ncol] = (_Float16)h;
            }
#pragma unroll
        for (int mt = 2; mt < MT_CNT; ++mt) {
            const int bsel = mt & 1;                   // tile mt-2 / mt-4
#pragma unroll
            for (int r = 0; r < 8; ++r)
                Aout[(mt * 16 + rof + r) * STRIDE + ncol] =
                    (_Float16)(th[bsel][r] * acc[mt][r]);
        }
    }
    __syncthreads();

    // ---- Stage C: heads (N=4) + closed-form 2x2 dynamics ------------------
    // 8 lanes per sample, 32-wide partial dots, shuffle-reduce.
    float zd0 = 0.f, zd1 = 0.f, zo = 0.f;
    float dzd0a = 0.f, dzd1a = 0.f, dzoa = 0.f, g0 = 0.f;   // tangent k=0
    float dzd0b = 0.f, dzd1b = 0.f, dzob = 0.f, g1 = 0.f;   // tangent k=1
    {
        const _Float16* h1r = Aout + s * STRIDE;
        const _Float16* t0r = Aout + (S_BLK + s) * STRIDE;
        const _Float16* t1r = Aout + (2 * S_BLK + s) * STRIDE;
        const int j0 = p * 32;
        for (int j = j0; j < j0 + 32; ++j) {
            const float wd0 = hWd0[j], wd1 = hWd1[j], wo = hWo[j], wv = hWv[j];
            const float h  = (float)h1r[j];
            zd0 = fmaf(h, wd0, zd0); zd1 = fmaf(h, wd1, zd1); zo = fmaf(h, wo, zo);
            const float d0 = (float)t0r[j];
            dzd0a = fmaf(d0, wd0, dzd0a); dzd1a = fmaf(d0, wd1, dzd1a);
            dzoa  = fmaf(d0, wo,  dzoa);  g0    = fmaf(d0, wv,  g0);
            const float d1 = (float)t1r[j];
            dzd0b = fmaf(d1, wd0, dzd0b); dzd1b = fmaf(d1, wd1, dzd1b);
            dzob  = fmaf(d1, wo,  dzob);  g1    = fmaf(d1, wv,  g1);
        }
    }
#define RED2(v) { v += __shfl_xor(v, 1, 32); v += __shfl_xor(v, 2, 32); v += __shfl_xor(v, 4, 32); }
    RED2(zd0); RED2(zd1); RED2(zo);
    RED2(dzd0a); RED2(dzd1a); RED2(dzoa); RED2(g0);
    RED2(dzd0b); RED2(dzd1b); RED2(dzob); RED2(g1);
#undef RED2

    if (valid && p == 0) {
        zd0 += bd[0]; zd1 += bd[1]; zo += bo[0];
        const float sg0 = 1.f / (1.f + expf(-zd0));    // softplus'
        const float sg1 = 1.f / (1.f + expf(-zd1));
        const float a = softplusf(zd0) + EPSV;         // L diag 0
        const float b = softplusf(zd1) + EPSV;         // L diag 1
        const float c = zo;                            // L off-diag
        // dL/dq_k entries
        const float da0 = sg0 * dzd0a, da1 = sg0 * dzd0b;
        const float db0 = sg1 * dzd1a, db1 = sg1 * dzd1b;
        const float dc0 = dzoa,        dc1 = dzob;
        // H = L L^T + eps I
        const float H00 = a * a + EPSV;
        const float H01 = a * c;
        const float H11 = b * b + c * c + EPSV;
        // dH/dq_k = dL L^T + L dL^T
        const float e00_0 = 2.f * a * da0,             e00_1 = 2.f * a * da1;
        const float e01_0 = da0 * c + a * dc0,         e01_1 = da1 * c + a * dc1;
        const float e11_0 = 2.f * (b * db0 + c * dc0), e11_1 = 2.f * (b * db1 + c * dc1);
        // hv_k = dH_k @ qd
        const float hv00 = e00_0 * qd0 + e01_0 * qd1;
        const float hv10 = e01_0 * qd0 + e11_0 * qd1;
        const float hv01 = e00_1 * qd0 + e01_1 * qd1;
        const float hv11 = e01_1 * qd0 + e11_1 * qd1;
        // KE_field_i = 0.5 qd^T dH_i qd
        const float ke0 = 0.5f * (qd0 * qd0 * e00_0 + 2.f * qd0 * qd1 * e01_0 + qd1 * qd1 * e11_0);
        const float ke1 = 0.5f * (qd0 * qd0 * e00_1 + 2.f * qd0 * qd1 * e01_1 + qd1 * qd1 * e11_1);
        const float C0 = hv00 * qd0 + hv01 * qd1 - ke0;
        const float C1 = hv10 * qd0 + hv11 * qd1 - ke1;
        const float r0 = tau0 - C0 - g0;
        const float r1 = tau1 - C1 - g1;
        const float inv = 1.f / (H00 * H11 - H01 * H01);
        out[gs * 6 + 0] = qd0;
        out[gs * 6 + 1] = qd1;
        out[gs * 6 + 2] = (H11 * r0 - H01 * r1) * inv;
        out[gs * 6 + 3] = (H00 * r1 - H01 * r0) * inv;
        out[gs * 6 + 4] = 0.f;
        out[gs * 6 + 5] = 0.f;
    }
}

extern "C" void kernel_launch(void* const* d_in, const int* in_sizes, int n_in,
                              void* d_out, int out_size, void* d_ws, size_t ws_size,
                              hipStream_t stream) {
    // setup_inputs order: t, x, W0, b0, W1, b1, Wd, bd, Wo, bo, Wv, bv
    const float* x  = (const float*)d_in[1];
    const float* W0 = (const float*)d_in[2];
    const float* b0 = (const float*)d_in[3];
    const float* W1 = (const float*)d_in[4];
    const float* b1 = (const float*)d_in[5];
    const float* Wd = (const float*)d_in[6];
    const float* bd = (const float*)d_in[7];
    const float* Wo = (const float*)d_in[8];
    const float* bo = (const float*)d_in[9];
    const float* Wv = (const float*)d_in[10];
    const int n = in_sizes[1] / 6;

    _Float16* W1sw = (_Float16*)d_ws;   // 128 KB of scratch

    const int shmem = 2 * ROWS * STRIDE * (int)sizeof(_Float16)
                    + 4 * NEU * (int)sizeof(float);           // 105,472 B

    // allow >64KB dynamic LDS (no-op if already permitted)
    static_cast<void>(hipFuncSetAttribute(
        (const void*)lode_dyn_kernel,
        hipFuncAttributeMaxDynamicSharedMemorySize, shmem));

    prep_w1_swizzle<<<256, 256, 0, stream>>>(W1, W1sw);

    const int blocks = (n + S_BLK - 1) / S_BLK;               // 4096 for n=131072
    lode_dyn_kernel<<<blocks, 256, (size_t)shmem, stream>>>(
        x, W0, b0, W1sw, b1, Wd, bd, Wo, bo, Wv, (float*)d_out, n);
}
